// IncrementalVGG16_18287970746562
// MI455X (gfx1250) — matrix operand order repaired and checked by reading, not compiled, not run on hardware
//
#include <hip/hip_runtime.h>
#include <hip/hip_bf16.h>

// ---------------------------------------------------------------------------
// VGG16 forward for MI455X (gfx1250).
//  - Activations: NHWC bf16 (channel-contiguous -> b128 vector loads/stores).
//    Input image channel-padded 3 -> 32 so ALL conv layers take the fast
//    WMMA path (Cin % 32 == 0); pad channels are zero, weights packed with
//    matching zeros, so results are unchanged.
//  - Convs: implicit GEMM, v_wmma_f32_16x16x32_bf16, K = (kh*3+kw)*Cin + ci.
//    Tap loop (kh,kw) fully unrolled; out-of-bounds taps redirect the lane's
//    B pointer to a zeroed pad page (no per-element masking, no exec branches).
//  - Weights pre-packed in WMMA-fragment-major layout (A = 1 v16u load/lane).
//  - FC: fp32, bandwidth-bound block-per-output reduction.
// ---------------------------------------------------------------------------

typedef __attribute__((ext_vector_type(16))) __bf16          v16bf;
typedef __attribute__((ext_vector_type(16))) unsigned short  v16u;
typedef __attribute__((ext_vector_type(8)))  unsigned short  v8u;
typedef __attribute__((ext_vector_type(8)))  float           v8f;

__device__ __forceinline__ unsigned short f2bf(float f) {
    unsigned int u = __float_as_uint(f);
    return (unsigned short)((u + 0x7FFFu + ((u >> 16) & 1u)) >> 16);   // RNE
}
__device__ __forceinline__ float bf2f(unsigned short h) {
    return __uint_as_float(((unsigned int)h) << 16);
}

// ---- zero the OOB pad page (ws is poisoned by the harness) ----------------
__global__ void zero_pad_kernel(unsigned int* __restrict__ p, int ndw) {
    int i = blockIdx.x * blockDim.x + threadIdx.x;
    if (i < ndw) p[i] = 0u;
}

// ---- input image: NCHW fp32 [8,3,224,224] -> NHWC bf16 [8,224,224,32] -----
__global__ void cvt_img_nhwc_kernel(const float* __restrict__ src,
                                    unsigned short* __restrict__ dst) {
    int i = blockIdx.x * blockDim.x + threadIdx.x;        // over 8*224*224*32
    if (i >= 8 * 224 * 224 * 32) return;
    int c  = i & 31;
    int t  = i >> 5;
    int w  = t % 224;
    int t2 = t / 224;
    int h  = t2 % 224;
    int b  = t2 / 224;
    unsigned short v = 0;
    if (c < 3)
        v = f2bf(src[(((size_t)b * 3 + c) * 224 + h) * 224 + w]);
    dst[i] = v;
}

// ---- conv weights: [O][CinReal][3][3] fp32 -> fragment-major packed bf16 --
// dst layout: [O/16][K/32][lane 0..31][slot 0..15], K = (kh*3+kw)*CinPad + ci.
// A-fragment slot map (contiguous k order, matches B loads): slot e:
//   kk = e + 16*(lane>=16);  m = lane&15.
__global__ void pack_wgt_kernel(const float* __restrict__ w,
                                unsigned short* __restrict__ dst,
                                int CinPad, int CinReal, int O, int K) {
    int i = blockIdx.x * blockDim.x + threadIdx.x;          // over O*K
    if (i >= O * K) return;
    int e    = i & 15;
    int lane = (i >> 4) & 31;
    int kt   = (i >> 9) % (K >> 5);
    int mt   = i / (32 * K / 2);
    int m    = mt * 16 + (lane & 15);
    int kk   = e + 16 * (lane >> 4);
    int k    = kt * 32 + kk;
    int p    = k / CinPad;
    int ci   = k - p * CinPad;
    int kh   = p / 3;
    int kw   = p - 3 * kh;
    unsigned short v = 0;
    if (ci < CinReal)
        v = f2bf(w[(((size_t)m * CinReal + ci) * 3 + kh) * 3 + kw]);
    dst[i] = v;
}

// ---- 2x2 maxpool, bf16 NHWC (coalesced: channel fastest) ------------------
__global__ void maxpool2_nhwc_kernel(const unsigned short* __restrict__ in,
                                     unsigned short* __restrict__ out,
                                     int C, int H, int W) {
    int Ho = H >> 1, Wo = W >> 1;
    int n = 8 * Ho * Wo * C;
    int i = blockIdx.x * blockDim.x + threadIdx.x;
    if (i >= n) return;
    int c  = i % C;
    int t  = i / C;
    int wo = t % Wo;
    int t2 = t / Wo;
    int ho = t2 % Ho;
    int b  = t2 / Ho;
    const unsigned short* p =
        in + (((size_t)b * H + 2 * ho) * W + 2 * wo) * C + c;
    float m = bf2f(p[0]);
    m = fmaxf(m, bf2f(p[C]));
    m = fmaxf(m, bf2f(p[(size_t)W * C]));
    m = fmaxf(m, bf2f(p[(size_t)W * C + C]));
    out[i] = f2bf(m);
}

// ---------------------------------------------------------------------------
// Implicit-GEMM 3x3 conv (pad=1, stride=1), NHWC bf16, f32 accumulate.
// Each wave: 16 out-channels x 64 pixels (4 accumulators, A reused 4x).
// Requires Cin % 32 == 0. Tap loop fully unrolled (kh,kw constant); per tap,
// each column's B pointer is either its input row or the zero pad page; the
// inner channel loop is pure pointer-increment b128 loads + WMMA.
// ---------------------------------------------------------------------------
__global__ __launch_bounds__(256)
void conv3x3_wmma_bf16_kernel(const unsigned short* __restrict__ in,
                              const unsigned short* __restrict__ wpk,
                              const float* __restrict__ bias,
                              unsigned short* __restrict__ out,
                              const unsigned short* __restrict__ zpad,
                              int Cin, int Cout, int H, int W, int wavesTotal) {
    const int lane = threadIdx.x & 31;
    const int wid  = blockIdx.x * (blockDim.x >> 5) + (threadIdx.x >> 5);
    if (wid >= wavesTotal) return;                 // wave-uniform exit

    const int K    = Cin * 9;                      // K % 32 == 0
    const int HW   = H * W;
    const int Npix = 8 * HW;
    const int n64  = (Npix + 63) >> 6;

    const int mt = wid / n64;
    const int nt = wid - mt * n64;
    const int m0 = mt * 16;
    const int n0 = nt * 64;

    const int hiHalf = lane >> 4;                  // 0 or 1
    const int kBbase = hiHalf ? 16 : 0;

    // Decode the 4 output-pixel columns this lane owns (once per wave).
    int cb[4], coh[4], cow[4];
    bool cok[4];
    #pragma unroll
    for (int t = 0; t < 4; ++t) {
        int colN = n0 + t * 16 + (lane & 15);
        bool ok  = colN < Npix;
        int p    = ok ? colN : 0;
        int b    = p / HW;
        int rem  = p - b * HW;
        int oh   = rem / W;
        cb[t] = b; coh[t] = oh; cow[t] = rem - oh * W; cok[t] = ok;
    }

    const v16u* wtile = (const v16u*)wpk + (size_t)mt * (K >> 5) * 32;
    v8f acc[4] = {v8f{}, v8f{}, v8f{}, v8f{}};

    // NHWC element offset of each column's center pixel.
    long pixOff[4];
    #pragma unroll
    for (int t = 0; t < 4; ++t)
        pixOff[t] = (((long)cb[t] * H + coh[t]) * W + cow[t]) * (long)Cin;

    int ktile = 0;                                  // wtile index, v16u units
    #pragma unroll
    for (int p = 0; p < 9; ++p) {
        const int kh = p / 3;                       // compile-time after unroll
        const int kw = p - 3 * kh;
        const long delta = ((long)(kh - 1) * W + (kw - 1)) * (long)Cin;

        // Per column: valid tap -> input row pointer, else zero page.
        const unsigned short* bp[4];
        #pragma unroll
        for (int t = 0; t < 4; ++t) {
            int ih = coh[t] + kh - 1;
            int iw = cow[t] + kw - 1;
            bool ok = cok[t] && (unsigned)ih < (unsigned)H
                             && (unsigned)iw < (unsigned)W;
            bp[t] = ok ? (in + pixOff[t] + delta) : zpad;   // 1 cndmask64
        }

        __builtin_prefetch(&wtile[(size_t)ktile + lane], 0, 1);
        for (int c0 = 0; c0 < Cin; c0 += 32) {
            v16u au = wtile[(size_t)ktile + lane];          // 2x b128
            ktile += 32;
            v16bf a = __builtin_bit_cast(v16bf, au);
            #pragma unroll
            for (int t = 0; t < 4; ++t) {
                v16u bv = *(const v16u*)(bp[t] + c0 + kBbase);  // 2x b128
                acc[t] = __builtin_amdgcn_wmma_f32_16x16x32_bf16(
                    false, a, false, __builtin_bit_cast(v16bf, bv),
                    (short)0, acc[t], false, false);
            }
        }
    }

    // Bias (8 consecutive channels for this lane's C rows).
    float bv[8];
    #pragma unroll
    for (int r = 0; r < 8; ++r) bv[r] = bias[m0 + 8 * hiHalf + r];

    // Store: C layout — VGPR r: M = r + 8*hiHalf, N = lane&15. NHWC makes the
    // 8 channels contiguous -> one packed b128 store per column tile.
    #pragma unroll
    for (int t = 0; t < 4; ++t) {
        if (!cok[t]) continue;
        v8u pv;
        #pragma unroll
        for (int r = 0; r < 8; ++r)
            pv[r] = f2bf(fmaxf(acc[t][r] + bv[r], 0.0f));
        size_t off = (((size_t)cb[t] * H + coh[t]) * W + cow[t]) * Cout
                     + m0 + 8 * hiHalf;
        *(v8u*)(out + off) = pv;
    }
}

// ---------------------------------------------------------------------------
// FC layer: out[b,o] = act( sum_k in[b,k] * w[o,k] + bias[o] )
// Bandwidth-bound on fp32 weights: block per output, 256 threads stride K,
// 8 batch accumulators, LDS tree reduction. NHWC remap handles the reference
// NCHW flatten of the bf16 [8,7,7,512] activation (k = c*49 + h*7 + w).
// ---------------------------------------------------------------------------
template <bool BF16_NHWC_IN, bool RELU>
__global__ __launch_bounds__(256)
void fc_kernel(const void* __restrict__ in_, const float* __restrict__ w,
               const float* __restrict__ bias, float* __restrict__ out,
               int K, int O) {
    __shared__ float red[8 * 256];
    const int o   = blockIdx.x;
    const int tid = threadIdx.x;
    const float* wrow = w + (size_t)o * K;
    const unsigned short* inb = (const unsigned short*)in_;
    const float* inf = (const float*)in_;

    float acc[8];
    #pragma unroll
    for (int b = 0; b < 8; ++b) acc[b] = 0.0f;

    for (int k = tid; k < K; k += 256) {
        float wv = wrow[k];
        size_t idx;
        if (BF16_NHWC_IN) {
            int c = k / 49;
            int r = k - 49 * c;
            int h = r / 7;
            int ww = r - 7 * h;
            idx = ((size_t)h * 7 + ww) * 512 + c;
        } else {
            idx = k;
        }
        #pragma unroll
        for (int b = 0; b < 8; ++b) {
            float xv = BF16_NHWC_IN
                     ? bf2f(inb[(size_t)b * 25088 + idx])
                     : inf[(size_t)b * K + k];
            acc[b] = fmaf(wv, xv, acc[b]);
        }
    }
    #pragma unroll
    for (int b = 0; b < 8; ++b) red[b * 256 + tid] = acc[b];
    __syncthreads();
    for (int s = 128; s > 0; s >>= 1) {
        if (tid < s) {
            #pragma unroll
            for (int b = 0; b < 8; ++b)
                red[b * 256 + tid] += red[b * 256 + tid + s];
        }
        __syncthreads();
    }
    if (tid < 8) {
        float v = red[tid * 256] + bias[o];
        if (RELU) v = fmaxf(v, 0.0f);
        out[(size_t)tid * O + o] = v;
    }
}

// ---------------------------------------------------------------------------
// Host orchestration
// ---------------------------------------------------------------------------
extern "C" void kernel_launch(void* const* d_in, const int* in_sizes, int n_in,
                              void* d_out, int out_size, void* d_ws, size_t ws_size,
                              hipStream_t stream) {
    (void)in_sizes; (void)n_in; (void)out_size; (void)ws_size;

    const float* x = (const float*)d_in[0];

    static const int  convC[13]     = {64,64,128,128,256,256,256,512,512,512,512,512,512};
    static const bool poolAfter[13] = {false,true,false,true,false,false,true,
                                       false,false,true,false,false,true};

    // ---- workspace carving ----
    const size_t ACT_BYTES   = (size_t)8 * 64 * 224 * 224 * 2;  // 51,380,224
    const size_t WPACK_ELEMS = 14727168;   // sum O*K over layers (layer1 Cin=32)
    const size_t ZPAD_ELEMS  = 1024;       // >= max Cin (2KB), zeroed
    char* base = (char*)d_ws;
    unsigned short* actA = (unsigned short*)(base);
    unsigned short* actB = (unsigned short*)(base + ACT_BYTES);
    unsigned short* wbuf = (unsigned short*)(base + 2 * ACT_BYTES);
    unsigned short* zpad = wbuf + WPACK_ELEMS;
    float* fcbuf0 = (float*)((char*)(zpad + ZPAD_ELEMS));
    float* fcbuf1 = fcbuf0 + (size_t)8 * 4096;

    // ---- zero pad page; convert input NCHW fp32 -> NHWC bf16 (C=32 pad) ----
    zero_pad_kernel<<<2, 256, 0, stream>>>((unsigned int*)zpad, (int)(ZPAD_ELEMS / 2));
    {
        int n = 8 * 224 * 224 * 32;
        cvt_img_nhwc_kernel<<<(n + 255) / 256, 256, 0, stream>>>(x, actA);
    }

    // ---- conv stack ----
    int H = 224;
    int inC = 3;                       // real input channels of next conv
    unsigned short* cur = actA;
    unsigned short* nxt = actB;
    size_t woff = 0;

    for (int ci = 0; ci < 13; ++ci) {
        const int O      = convC[ci];
        const int cinPad = (ci == 0) ? 32 : inC;   // layer 1 channel-padded
        const int K      = cinPad * 9;             // K % 32 == 0 for all layers
        const float* wf = (const float*)d_in[4 + 2 * ci];
        const float* bf = (const float*)d_in[5 + 2 * ci];
        unsigned short* wb = wbuf + woff;
        woff += (size_t)O * K;

        {
            int n = O * K;
            pack_wgt_kernel<<<(n + 255) / 256, 256, 0, stream>>>(wf, wb,
                                                                 cinPad, inC, O, K);
        }

        const int Npix = 8 * H * H;
        const int n64  = (Npix + 63) / 64;
        const int wavesTotal = (O / 16) * n64;
        const int blocks = (wavesTotal + 7) / 8;      // 8 waves / 256-thread block
        conv3x3_wmma_bf16_kernel<<<blocks, 256, 0, stream>>>(cur, wb, bf, nxt, zpad,
                                                             cinPad, O, H, H, wavesTotal);
        { unsigned short* t = cur; cur = nxt; nxt = t; }
        inC = O;

        if (poolAfter[ci]) {
            int n = 8 * (H / 2) * (H / 2) * inC;
            maxpool2_nhwc_kernel<<<(n + 255) / 256, 256, 0, stream>>>(cur, nxt, inC, H, H);
            { unsigned short* t = cur; cur = nxt; nxt = t; }
            H /= 2;
        }
    }

    // ---- FC stack: cur = bf16 NHWC [8,7,7,512]; reference flatten is NCHW ----
    const float* fw0 = (const float*)d_in[30];
    const float* fb0 = (const float*)d_in[31];
    const float* fw1 = (const float*)d_in[32];
    const float* fb1 = (const float*)d_in[33];
    const float* fw2 = (const float*)d_in[34];
    const float* fb2 = (const float*)d_in[35];

    fc_kernel<true,  true ><<<4096, 256, 0, stream>>>(cur,    fw0, fb0, fcbuf0, 25088, 4096);
    fc_kernel<false, true ><<<4096, 256, 0, stream>>>(fcbuf0, fw1, fb1, fcbuf1, 4096, 4096);
    fc_kernel<false, false><<<1000, 256, 0, stream>>>(fcbuf1, fw2, fb2, (float*)d_out, 4096, 1000);
}
